// Linear4Bit_14224931684655
// MI455X (gfx1250) — compile-verified, matrix operand order checked by reading, not run on hardware
//
#include <hip/hip_runtime.h>

typedef __attribute__((ext_vector_type(16))) _Float16 v16h;
typedef __attribute__((ext_vector_type(8)))  float    v8f;
typedef __attribute__((ext_vector_type(4)))  float    f32x4;
typedef __attribute__((ext_vector_type(4)))  int      i32x4;
typedef __attribute__((ext_vector_type(4)))  unsigned int u32x4;

#define TOKENS 4096
#define IN_F   4096
#define OUT_F  11008
#define GROUPS_PER_ROW (IN_F / 16)
#define TOTAL_GROUPS (OUT_F * GROUPS_PER_ROW)

#define BT 256          // tokens per block (M tile)
#define BO 128          // out-features per block (N tile)
#define BK 32           // K per step (WMMA f16 K)
#define LDS_STRIDE 40   // halves per row (80B; multiple of 16B, spreads banks)

#define X16_BYTES ((size_t)TOKENS * IN_F * 2)          // 32 MB
#define W16_BYTES ((size_t)OUT_F * IN_F * 2)           // 90 MB

// ---------------------------------------------------------------------------
// Prepass 1: x fp32 -> f16 (once; removes per-block re-conversion)
// ---------------------------------------------------------------------------
__global__ __launch_bounds__(256)
void cvt_x_kernel(const float* __restrict__ x, _Float16* __restrict__ x16)
{
    size_t idx = ((size_t)blockIdx.x * 256 + threadIdx.x) * 8;
    const f32x4* p = (const f32x4*)(x + idx);
    f32x4 v0 = p[0];
    f32x4 v1 = p[1];
    union { _Float16 h[8]; u32x4 u; } q;
    q.h[0] = (_Float16)v0.x; q.h[1] = (_Float16)v0.y;
    q.h[2] = (_Float16)v0.z; q.h[3] = (_Float16)v0.w;
    q.h[4] = (_Float16)v1.x; q.h[5] = (_Float16)v1.y;
    q.h[6] = (_Float16)v1.z; q.h[7] = (_Float16)v1.w;
    *(u32x4*)(x16 + idx) = q.u;
}

// ---------------------------------------------------------------------------
// Prepass 2: dequantize W once -> f16 row-major [OUT_F][IN_F]
// ---------------------------------------------------------------------------
__global__ __launch_bounds__(256)
void deq_w_kernel(const int* __restrict__ wq, const _Float16* __restrict__ wn,
                  _Float16* __restrict__ w16)
{
    int g = blockIdx.x * 256 + threadIdx.x;        // one 16-wide group / thread
    int o  = g >> 8;
    int gi = g & 255;
    const i32x4* qp = (const i32x4*)(wq + (size_t)g * 8);
    i32x4 q0 = qp[0];
    i32x4 q1 = qp[1];
    float norm  = (float)wn[g];
    float scale = norm * (2.0f / 15.0f);
    int qw[8] = { q0.x, q0.y, q0.z, q0.w, q1.x, q1.y, q1.z, q1.w };
    union { _Float16 h[16]; u32x4 u[2]; } d;
    #pragma unroll
    for (int j = 0; j < 8; ++j) {
        int lo =  qw[j]       & 15;
        int hi = (qw[j] >> 4) & 15;
        d.h[2*j]   = (_Float16)(scale * (float)lo - norm);
        d.h[2*j+1] = (_Float16)(scale * (float)hi - norm);
    }
    u32x4* op = (u32x4*)(w16 + (size_t)o * IN_F + gi * 16);
    op[0] = d.u[0];
    op[1] = d.u[1];
}

// ---------------------------------------------------------------------------
// Main GEMM (prepass path): async global->LDS staging, double-buffered
// ---------------------------------------------------------------------------
__global__ __launch_bounds__(256)
void gemm_wmma_pre(const _Float16* __restrict__ x16,
                   const _Float16* __restrict__ w16,
                   const float* __restrict__ bias,
                   float* __restrict__ out)
{
    __shared__ _Float16 Xs[2][BT * LDS_STRIDE];   // 2 x 20480 B
    __shared__ _Float16 Ws[2][BO * LDS_STRIDE];   // 2 x 10240 B

    const int tid  = threadIdx.x;
    const int lane = tid & 31;
    const int wave = tid >> 5;

    const int row0 = blockIdx.y * BT;
    const int col0 = blockIdx.x * BO;

    const int waveM = (wave >> 1) * 64;
    const int waveN = (wave & 1) * 64;

    v8f acc[4][4] = {};

    const int wcol = tid >> 1;
    const int wh   = tid & 1;

    const int arow = waveM + (lane & 15);
    const int brow = waveN + (lane & 15);
    const int koff = (lane >> 4) * 16;

    // LDS byte addresses (generic ptr low 32 bits == LDS offset)
    const unsigned xs0 = (unsigned)(unsigned long long)(void*)&Xs[0][0];
    const unsigned ws0 = (unsigned)(unsigned long long)(void*)&Ws[0][0];

    // issue 6 async 16B copies per thread: 4 for X tile, 2 for W tile
    auto issue_async = [&](int kk, int buf) {
        #pragma unroll
        for (int i = 0; i < 4; ++i) {
            int idx = tid + i * 256;            // 1024 16B chunks of X tile
            int r   = idx >> 2;
            int c8  = idx & 3;
            unsigned lds = xs0 + (unsigned)buf * (BT * LDS_STRIDE * 2)
                         + (unsigned)(r * (LDS_STRIDE * 2) + c8 * 16);
            unsigned long long ga = (unsigned long long)
                (x16 + (size_t)(row0 + r) * IN_F + kk + c8 * 8);
            asm volatile("global_load_async_to_lds_b128 %0, %1, off"
                         :: "v"(lds), "v"(ga) : "memory");
        }
        {
            unsigned lds = ws0 + (unsigned)buf * (BO * LDS_STRIDE * 2)
                         + (unsigned)(wcol * (LDS_STRIDE * 2) + wh * 32);
            unsigned long long ga = (unsigned long long)
                (w16 + (size_t)(col0 + wcol) * IN_F + kk + wh * 16);
            asm volatile("global_load_async_to_lds_b128 %0, %1, off"
                         :: "v"(lds), "v"(ga) : "memory");
            asm volatile("global_load_async_to_lds_b128 %0, %1, off"
                         :: "v"(lds + 16u), "v"(ga + 16ull) : "memory");
        }
    };

    issue_async(0, 0);

    int cur = 0;
    for (int kk = 0; kk < IN_F; kk += BK, cur ^= 1) {
        // my async copies into buf[cur] have landed
        asm volatile("s_wait_asynccnt 0x0" ::: "memory");
        // all waves' copies landed; also fences iter i-1 readers of buf[cur^1]
        __syncthreads();

        if (kk + BK < IN_F)
            issue_async(kk + BK, cur ^ 1);   // runs under the WMMA burst

        v16h bfrag[4], afrag[4];
        #pragma unroll
        for (int ni = 0; ni < 4; ++ni) {
            union { u32x4 q[2]; v16h h; } u;
            const u32x4* p = (const u32x4*)(&Ws[cur][(brow + ni * 16) * LDS_STRIDE + koff]);
            u.q[0] = p[0]; u.q[1] = p[1];
            bfrag[ni] = u.h;
        }
        #pragma unroll
        for (int mi = 0; mi < 4; ++mi) {
            union { u32x4 q[2]; v16h h; } u;
            const u32x4* p = (const u32x4*)(&Xs[cur][(arow + mi * 16) * LDS_STRIDE + koff]);
            u.q[0] = p[0]; u.q[1] = p[1];
            afrag[mi] = u.h;
        }

        #pragma unroll
        for (int mi = 0; mi < 4; ++mi)
            #pragma unroll
            for (int ni = 0; ni < 4; ++ni)
                acc[mi][ni] = __builtin_amdgcn_wmma_f32_16x16x32_f16(
                    false, afrag[mi], false, bfrag[ni],
                    (short)0, acc[mi][ni], false, false);
    }

    const int colBase = col0 + waveN + (lane & 15);
    const int rowHalf = (lane >> 4) * 8;
    #pragma unroll
    for (int ni = 0; ni < 4; ++ni) {
        int gcol = colBase + ni * 16;
        float bv = bias[gcol];
        #pragma unroll
        for (int mi = 0; mi < 4; ++mi) {
            #pragma unroll
            for (int r = 0; r < 8; ++r) {
                int grow = row0 + waveM + mi * 16 + rowHalf + r;
                out[(size_t)grow * OUT_F + gcol] = acc[mi][ni][r] + bv;
            }
        }
    }
}

// ---------------------------------------------------------------------------
// Fallback: fused dequant GEMM (ws-free)
// ---------------------------------------------------------------------------
__global__ __launch_bounds__(256)
void gemm_wmma_fused(const float* __restrict__ x,
                     const int* __restrict__ wq,
                     const _Float16* __restrict__ wn,
                     const float* __restrict__ bias,
                     float* __restrict__ out)
{
    __shared__ _Float16 Xs[2][BT * LDS_STRIDE];
    __shared__ _Float16 Ws[2][BO * LDS_STRIDE];

    const int tid  = threadIdx.x;
    const int lane = tid & 31;
    const int wave = tid >> 5;

    const int row0 = blockIdx.y * BT;
    const int col0 = blockIdx.x * BO;

    const int waveM = (wave >> 1) * 64;
    const int waveN = (wave & 1) * 64;

    v8f acc[4][4] = {};

    const int wcol = tid >> 1;
    const int wh   = tid & 1;

    const int arow = waveM + (lane & 15);
    const int brow = waveN + (lane & 15);
    const int koff = (lane >> 4) * 16;

    u32x4 xreg[4];
    u32x4 wreg[2];

    auto load_x = [&](int kk) {
        #pragma unroll
        for (int i = 0; i < 4; ++i) {
            int idx = tid + i * 256;
            int r   = idx >> 2;
            int c8  = idx & 3;
            const f32x4* p = (const f32x4*)(x + (size_t)(row0 + r) * IN_F + kk + c8 * 8);
            f32x4 v0 = p[0];
            f32x4 v1 = p[1];
            union { _Float16 h[8]; u32x4 u; } q;
            q.h[0] = (_Float16)v0.x; q.h[1] = (_Float16)v0.y;
            q.h[2] = (_Float16)v0.z; q.h[3] = (_Float16)v0.w;
            q.h[4] = (_Float16)v1.x; q.h[5] = (_Float16)v1.y;
            q.h[6] = (_Float16)v1.z; q.h[7] = (_Float16)v1.w;
            xreg[i] = q.u;
        }
    };
    auto load_w = [&](int kk) {
        int g = (col0 + wcol) * GROUPS_PER_ROW + (kk >> 4) + wh;
        const i32x4* qp = (const i32x4*)(wq + (size_t)g * 8);
        i32x4 q0 = qp[0];
        i32x4 q1 = qp[1];
        float norm  = (float)wn[g];
        float scale = norm * (2.0f / 15.0f);
        int qw[8] = { q0.x, q0.y, q0.z, q0.w, q1.x, q1.y, q1.z, q1.w };
        union { _Float16 h[16]; u32x4 u[2]; } d;
        #pragma unroll
        for (int j = 0; j < 8; ++j) {
            int lo =  qw[j]       & 15;
            int hi = (qw[j] >> 4) & 15;
            d.h[2*j]   = (_Float16)(scale * (float)lo - norm);
            d.h[2*j+1] = (_Float16)(scale * (float)hi - norm);
        }
        wreg[0] = d.u[0];
        wreg[1] = d.u[1];
    };

    load_x(0);
    load_w(0);

    int cur = 0;
    for (int kk = 0; kk < IN_F; kk += BK, cur ^= 1) {
        #pragma unroll
        for (int i = 0; i < 4; ++i) {
            int idx = tid + i * 256;
            int r   = idx >> 2;
            int c8  = idx & 3;
            *(u32x4*)(&Xs[cur][r * LDS_STRIDE + c8 * 8]) = xreg[i];
        }
        {
            u32x4* wp = (u32x4*)(&Ws[cur][wcol * LDS_STRIDE + wh * 16]);
            wp[0] = wreg[0];
            wp[1] = wreg[1];
        }
        __syncthreads();

        if (kk + BK < IN_F) {
            load_x(kk + BK);
            load_w(kk + BK);
        }

        v16h bfrag[4], afrag[4];
        #pragma unroll
        for (int ni = 0; ni < 4; ++ni) {
            union { u32x4 q[2]; v16h h; } u;
            const u32x4* p = (const u32x4*)(&Ws[cur][(brow + ni * 16) * LDS_STRIDE + koff]);
            u.q[0] = p[0]; u.q[1] = p[1];
            bfrag[ni] = u.h;
        }
        #pragma unroll
        for (int mi = 0; mi < 4; ++mi) {
            union { u32x4 q[2]; v16h h; } u;
            const u32x4* p = (const u32x4*)(&Xs[cur][(arow + mi * 16) * LDS_STRIDE + koff]);
            u.q[0] = p[0]; u.q[1] = p[1];
            afrag[mi] = u.h;
        }

        #pragma unroll
        for (int mi = 0; mi < 4; ++mi)
            #pragma unroll
            for (int ni = 0; ni < 4; ++ni)
                acc[mi][ni] = __builtin_amdgcn_wmma_f32_16x16x32_f16(
                    false, afrag[mi], false, bfrag[ni],
                    (short)0, acc[mi][ni], false, false);
    }

    const int colBase = col0 + waveN + (lane & 15);
    const int rowHalf = (lane >> 4) * 8;
    #pragma unroll
    for (int ni = 0; ni < 4; ++ni) {
        int gcol = colBase + ni * 16;
        float bv = bias[gcol];
        #pragma unroll
        for (int mi = 0; mi < 4; ++mi) {
            #pragma unroll
            for (int r = 0; r < 8; ++r) {
                int grow = row0 + waveM + mi * 16 + rowHalf + r;
                out[(size_t)grow * OUT_F + gcol] = acc[mi][ni][r] + bv;
            }
        }
    }
}

extern "C" void kernel_launch(void* const* d_in, const int* in_sizes, int n_in,
                              void* d_out, int out_size, void* d_ws, size_t ws_size,
                              hipStream_t stream) {
    const float*    x    = (const float*)d_in[0];
    const int*      wq   = (const int*)d_in[1];
    const _Float16* wn   = (const _Float16*)d_in[2];
    const float*    bias = (const float*)d_in[3];
    float*          out  = (float*)d_out;

    dim3 grid(OUT_F / BO, TOKENS / BT);   // 86 x 16 workgroups

    if (ws_size >= X16_BYTES + W16_BYTES) {
        _Float16* x16 = (_Float16*)d_ws;
        _Float16* w16 = (_Float16*)((char*)d_ws + X16_BYTES);
        cvt_x_kernel<<<(TOKENS * IN_F) / (256 * 8), 256, 0, stream>>>(x, x16);
        deq_w_kernel<<<TOTAL_GROUPS / 256, 256, 0, stream>>>(wq, wn, w16);
        gemm_wmma_pre<<<grid, 256, 0, stream>>>(x16, w16, bias, out);
    } else {
        gemm_wmma_fused<<<grid, 256, 0, stream>>>(x, wq, wn, bias, out);
    }
}
